// LocalSphericalNeuralOperatorNet_39333310497240
// MI455X (gfx1250) — matrix operand order repaired and checked by reading, not compiled, not run on hardware
//
#include <hip/hip_runtime.h>

#define PI_F 3.14159265358979323846f

typedef __attribute__((ext_vector_type(16))) __bf16 v16bf;
typedef __attribute__((ext_vector_type(8)))  float  v8f;

// ---------------- bf16 helpers (RNE) ----------------
__device__ __forceinline__ __bf16 f2bf(float f) {
  union { float f; unsigned u; } v; v.f = f;
  unsigned r = v.u + 0x7fffu + ((v.u >> 16) & 1u);
  union { unsigned short s; __bf16 b; } o; o.s = (unsigned short)(r >> 16);
  return o.b;
}

// ---------------- generic strided batched WMMA GEMM ----------------
// C[z][m][n] (+)= (+/-) sum_k A[z][m][k] * B[z][k][n]  (+ bias[n]) (relu)
// element addr = (z/bdiv)*hi + (z%bdiv)*lo + row*rs + col*cs
struct OpS { const float* p; long long hi, lo, rs, cs; };
struct GemmP {
  OpS A, B;
  float* C; long long cHi, cLo, cRs, cCs;
  const float* bias;
  int M, N, K, bdiv, acc, neg, relu;
};

#define BM 64
#define BN 64
#define BK 32

__global__ __launch_bounds__(128) void k_gemm(GemmP p) {
  __shared__ __bf16 As[BM][BK + 4];
  __shared__ __bf16 Bs[BK][BN + 4];

  const int z   = blockIdx.z;
  const int zh  = z / p.bdiv, zl = z % p.bdiv;
  const long long za = (long long)zh * p.A.hi + (long long)zl * p.A.lo;
  const long long zb = (long long)zh * p.B.hi + (long long)zl * p.B.lo;
  const long long zc = (long long)zh * p.cHi  + (long long)zl * p.cLo;

  const int m0 = blockIdx.x * BM;
  const int n0 = blockIdx.y * BN;
  const int tid = threadIdx.x;
  const int wave = tid >> 5, lane = tid & 31;
  const int wm = (wave >> 1) * 32, wn = (wave & 1) * 32;
  const int laneHi = lane >> 4, lane16 = lane & 15;

  v8f zerov = {0.f,0.f,0.f,0.f,0.f,0.f,0.f,0.f};
  v8f acc[2][2] = {{zerov, zerov}, {zerov, zerov}};

  for (int k0 = 0; k0 < p.K; k0 += BK) {
    // stage A tile (BM x BK)
    for (int t = tid; t < BM * BK; t += 128) {
      int r = t >> 5, c = t & 31;
      int gr = m0 + r, gk = k0 + c;
      float v = 0.f;
      if (gr < p.M && gk < p.K) {
        long long offA = za + (long long)gr * p.A.rs + (long long)gk * p.A.cs;
        v = p.A.p[offA];
        if (gk + BK < p.K) __builtin_prefetch(p.A.p + offA + (long long)BK * p.A.cs, 0, 1);
      }
      As[r][c] = f2bf(v);
    }
    // stage B tile (BK x BN)
    for (int t = tid; t < BK * BN; t += 128) {
      int r = t >> 6, c = t & 63;
      int gk = k0 + r, gn = n0 + c;
      float v = 0.f;
      if (gk < p.K && gn < p.N) {
        long long offB = zb + (long long)gk * p.B.rs + (long long)gn * p.B.cs;
        v = p.B.p[offB];
        if (gk + BK < p.K) __builtin_prefetch(p.B.p + offB + (long long)BK * p.B.rs, 0, 1);
      }
      Bs[r][c] = f2bf(v);
    }
    __syncthreads();

    #pragma unroll
    for (int sm = 0; sm < 2; ++sm) {
      #pragma unroll
      for (int sn = 0; sn < 2; ++sn) {
        v16bf a, b;
        const int mrow = wm + sm * 16 + lane16;
        const int ncol = wn + sn * 16 + lane16;
        #pragma unroll
        for (int v = 0; v < 8; ++v) {
          // A 16x32 bf16 layout: V0..3 -> K 0..7 (lanes 0-15) / 8..15 (16-31); V4..7 -> +16
          int kA = ((v >> 2) << 4) + (laneHi << 3) + ((v & 3) << 1);
          a[2 * v]     = As[mrow][kA];
          a[2 * v + 1] = As[mrow][kA + 1];
          // B 32x16 bf16 layout: V0..3 -> K 0..7 (lanes 0-15) / 8..15 (16-31); V4..7 -> +16
          int kB = ((v >> 2) << 4) + (laneHi << 3) + ((v & 3) << 1);
          b[2 * v]     = Bs[kB][ncol];
          b[2 * v + 1] = Bs[kB + 1][ncol];
        }
        acc[sm][sn] = __builtin_amdgcn_wmma_f32_16x16x32_bf16(
            false, a, false, b, (short)0, acc[sm][sn], false, false);
      }
    }
    __syncthreads();
  }

  // epilogue: D 16x16 f32 layout: VGPR r holds M = r + 8*(lane/16), N = lane%16
  #pragma unroll
  for (int sm = 0; sm < 2; ++sm)
  #pragma unroll
  for (int sn = 0; sn < 2; ++sn)
  #pragma unroll
  for (int r = 0; r < 8; ++r) {
    int row = m0 + wm + sm * 16 + r + (laneHi << 3);
    int col = n0 + wn + sn * 16 + lane16;
    if (row < p.M && col < p.N) {
      float v = acc[sm][sn][r];
      if (p.neg) v = -v;
      if (p.bias) v += p.bias[col];
      long long offC = zc + (long long)row * p.cRs + (long long)col * p.cCs;
      if (p.acc) v += p.C[offC];
      if (p.relu) v = v > 0.f ? v : 0.f;
      p.C[offC] = v;
    }
  }
}

// ---------------- DFT / iDFT tables (exact integer phase reduction) ----------------
__global__ void k_dft_fwd(float* C, float* S, int n, int M) {
  int i = blockIdx.x * 256 + threadIdx.x;
  if (i >= n * M) return;
  int m = i % M, w = i / M;
  int ph = (int)(((long long)w * m) % n);
  float ang = 2.f * PI_F * ph / n;
  C[i] = cosf(ang);
  S[i] = sinf(ang);
}

__global__ void k_dft_inv(float* IC, float* IS, int n, int M, int stride, int Wout) {
  int i = blockIdx.x * 256 + threadIdx.x;
  if (i >= M * Wout) return;
  int w = i % Wout, m = i / Wout;
  float a = (m == 0 || 2 * m == n) ? 1.f : 2.f;
  int ph = (int)(((long long)m * (w * stride)) % n);
  float ang = 2.f * PI_F * ph / n;
  IC[i] =  a * cosf(ang) / n;
  IS[i] = -a * sinf(ang) / n;
}

// ---------------- normalized associated Legendre (PT [l][m][h], PW = PT*quad) ----------------
__global__ void k_legendre(float* PT, float* PW) {
  int h = threadIdx.x;
  if (h >= 32) return;
  float theta = PI_F * (h + 0.5f) / 32.f;
  float ct = cosf(theta), st = sinf(theta);
  auto idx = [&](int l, int m) { return (l * 32 + m) * 32 + h; };
  for (int l = 0; l < 32; ++l)
    for (int m = 0; m < 32; ++m) PT[idx(l, m)] = 0.f;
  float pmm = 1.f / sqrtf(2.f);
  PT[idx(0, 0)] = pmm;
  for (int m = 1; m < 32; ++m) {
    pmm = sqrtf((2.f * m + 1.f) / (2.f * m)) * st * pmm;
    PT[idx(m, m)] = pmm;
  }
  for (int m = 0; m < 31; ++m)
    PT[idx(m + 1, m)] = sqrtf(2.f * m + 3.f) * ct * PT[idx(m, m)];
  for (int m = 0; m < 32; ++m)
    for (int l = m + 2; l < 32; ++l) {
      float a = sqrtf((4.f * l * l - 1.f) / ((float)l * l - (float)m * m));
      float b = a * sqrtf(((l - 1.f) * (l - 1.f) - m * m) / (4.f * (l - 1.f) * (l - 1.f) - 1.f));
      PT[idx(l, m)] = a * ct * PT[idx(l - 1, m)] - b * PT[idx(l - 2, m)];
    }
  float q = st * (PI_F / 32.f);
  for (int l = 0; l < 32; ++l)
    for (int m = 0; m < 32; ++m) PW[idx(l, m)] = PT[idx(l, m)] * q;
}

// ---------------- DISCO spatial psi builder (K=9 basis per point) ----------------
__global__ void k_psi(float* outp, int nlat_in, int nlon, int nlat_out, float cutoff,
                      int ho0, int nho, int k0, int nk,
                      long long sk, long long sho, long long shi) {
  long long i = (long long)blockIdx.x * 256 + threadIdx.x;
  long long n = (long long)nho * nlat_in * nlon;
  if (i >= n) return;
  int w = (int)(i % nlon); long long t = i / nlon;
  int hi = (int)(t % nlat_in);
  int hol = (int)(t / nlat_in);
  float ti = PI_F * (hi + 0.5f) / nlat_in;
  float to = PI_F * (ho0 + hol + 0.5f) / nlat_out;
  float ph = 2.f * PI_F * w / nlon;
  float cto = cosf(to), sto = sinf(to), cti = cosf(ti), sti = sinf(ti);
  float cph = cosf(ph), sph = sinf(ph);
  float cr = cto * cti + sto * sti * cph;
  cr = fminf(fmaxf(cr, -1.f), 1.f);
  float r = acosf(cr);
  float alpha = atan2f(sti * sph, cti * sto - sti * cto * cph);
  float quad = sti * (PI_F / nlat_in) * (2.f * PI_F / nlon) * (r <= cutoff ? 1.f : 0.f);
  float dr = cutoff * 0.5f;         // NR-1 = 2
  float dphi = 0.5f * PI_F;         // 2*pi / NPHI
  float vals[9];
  vals[0] = fminf(fmaxf(1.f - r / dr, 0.f), 1.f) * quad;
  for (int j = 1; j <= 2; ++j) {
    float tr = fminf(fmaxf(1.f - fabsf(r - j * dr) / dr, 0.f), 1.f);
    for (int kk = 0; kk < 4; ++kk) {
      float m = fmodf(alpha - kk * dphi + PI_F, 2.f * PI_F);
      if (m < 0.f) m += 2.f * PI_F;
      float d = fabsf(m - PI_F);
      vals[1 + (j - 1) * 4 + kk] = tr * fminf(fmaxf(1.f - d / dphi, 0.f), 1.f) * quad;
    }
  }
  long long base = (long long)hol * sho + (long long)hi * shi + w;
  for (int k = k0; k < k0 + nk; ++k)
    outp[(long long)(k - k0) * sk + base] = vals[k];
}

// ---------------- generic 5-D permute (input contiguous [d0,d1,d2,d3,d4]) ----------------
__global__ void k_permute(const float* in, float* out, long long n,
                          int d1, int d2, int d3, int d4,
                          long long s0, long long s1, long long s2, long long s3, long long s4) {
  long long i = (long long)blockIdx.x * 256 + threadIdx.x;
  if (i >= n) return;
  long long t = i;
  int i4 = (int)(t % d4); t /= d4;
  int i3 = (int)(t % d3); t /= d3;
  int i2 = (int)(t % d2); t /= d2;
  int i1 = (int)(t % d1); t /= d1;
  long long i0 = t;
  out[i0 * s0 + (long long)i1 * s1 + (long long)i2 * s2 + (long long)i3 * s3 + (long long)i4 * s4] = in[i];
}

// ---------------- bilinear resample (32,64)->(128,256), periodic in lon ----------------
// v: [b, h32, w64, ok27] (pixel-major)  ->  u: [b][ok][128][256]
__global__ void k_resample(const float* v, float* u) {
  long long nTot = 8ll * 27 * 128 * 256;
  long long i = (long long)blockIdx.x * 256 + threadIdx.x;
  if (i >= nTot) return;
  int wo = (int)(i % 256); long long t = i / 256;
  int ho = (int)(t % 128); t /= 128;
  int ok = (int)(t % 27);
  int b  = (int)(t / 27);
  float fi = (ho + 0.5f) * 32.f / 128.f - 0.5f;
  fi = fminf(fmaxf(fi, 0.f), 31.f);
  int i0 = (int)floorf(fi);
  int i1 = i0 + 1 > 31 ? 31 : i0 + 1;
  float wl = fi - (float)i0;
  float fj = wo * 0.25f;
  int j0 = (int)floorf(fj);
  int j1 = (j0 + 1) & 63;
  float wj = fj - (float)j0;
  j0 &= 63;
  auto V = [&](int hh, int ww) {
    return v[(((long long)b * 32 + hh) * 64 + ww) * 27 + ok];
  };
  float a0 = V(i0, j0) * (1.f - wl) + V(i1, j0) * wl;
  float a1 = V(i0, j1) * (1.f - wl) + V(i1, j1) * wl;
  u[(((long long)b * 27 + ok) * 128 + ho) * 256 + wo] = a0 * (1.f - wj) + a1 * wj;
}

// ==================== host orchestration ====================
static inline OpS OP(const float* p, long long hi, long long lo, long long rs, long long cs) {
  OpS o; o.p = p; o.hi = hi; o.lo = lo; o.rs = rs; o.cs = cs; return o;
}

static void gemmL(hipStream_t st, int M, int N, int K, int nb, int bdiv,
                  OpS A, OpS B, float* C,
                  long long cHi, long long cLo, long long cRs, long long cCs,
                  const float* bias, int accF, int negF, int reluF) {
  GemmP p;
  p.A = A; p.B = B; p.C = C; p.cHi = cHi; p.cLo = cLo; p.cRs = cRs; p.cCs = cCs;
  p.bias = bias; p.M = M; p.N = N; p.K = K;
  p.bdiv = bdiv < 1 ? 1 : bdiv;
  p.acc = accF; p.neg = negF; p.relu = reluF;
  dim3 grid((M + BM - 1) / BM, (N + BN - 1) / BN, nb < 1 ? 1 : nb);
  k_gemm<<<grid, dim3(128), 0, st>>>(p);
}

static inline unsigned gb(long long n) { return (unsigned)((n + 255) / 256); }

extern "C" void kernel_launch(void* const* d_in, const int* in_sizes, int n_in,
                              void* d_out, int out_size, void* d_ws, size_t ws_size,
                              hipStream_t stream) {
  (void)in_sizes; (void)n_in; (void)out_size; (void)ws_size;
  const float* x    = (const float*)d_in[0];
  const float* encw = (const float*)d_in[1];
  const float* lconv= (const float*)d_in[2];
  const float* gconv= (const float*)d_in[3];
  const float* mw1  = (const float*)d_in[4];
  const float* mb1  = (const float*)d_in[5];
  const float* mw2  = (const float*)d_in[6];
  const float* mb2  = (const float*)d_in[7];
  const float* skw  = (const float*)d_in[8];
  const float* skb  = (const float*)d_in[9];
  const float* decw = (const float*)d_in[10];
  float* out = (float*)d_out;

  float* Wp = (float*)d_ws;
  long long off = 0;
  auto al = [&](long long n) { float* r = Wp + off; off += (n + 63) & ~63ll; return r; };

  // ---- tables ----
  float *C256 = al(33024), *S256 = al(33024), *IC256 = al(33024), *IS256 = al(33024);
  float *ICse = al(8256),  *ISse = al(8256);
  float *C64  = al(2112),  *S64  = al(2112),  *IC64 = al(2112), *IS64 = al(2112);
  float *PWt  = al(32768), *PTt  = al(32768);
  float *PencR= al(4755456), *PencI = al(4755456);
  float *PintR= al(304128),  *PintI = al(304128);
  float *SPAT = al(4718592);
  float *DWP  = al(6912);
  // ---- activations ----
  float *Xr256= al(396288), *Xi256 = al(396288);
  float *ZeR  = al(891648), *ZeI   = al(891648);
  float *Zt   = al(442368), *ZP    = al(442368);
  float *A0b  = al(4194304), *A1b = al(4194304), *A2b = al(4194304);
  float *H1b  = al(8388608);
  float *XPb  = al(4194304);
  float *Xr64 = al(2162688), *Xi64 = al(2162688);
  float *Z2R  = al(1769472), *Z2I  = al(1769472);
  float *YhR  = al(2162688), *YhI  = al(2162688);
  float *crB  = al(2097152), *ciB  = al(2097152), *c2R = al(2097152), *c2I = al(2097152);
  float *vb   = al(442368),  *ub   = al(7077888);
  float *UR   = al(3566592), *UI   = al(3566592);
  float *PDR  = al(2377728), *PDI  = al(2377728);
  float *YdR  = al(396288),  *YdI  = al(396288);
  float *XoR = crB, *XoI = ciB;   // reuse: cr/ci dead before synthesis

  // ---- build constant operators ----
  k_dft_fwd<<<gb(33024), 256, 0, stream>>>(C256, S256, 256, 129);
  k_dft_fwd<<<gb(2112), 256, 0, stream>>>(C64, S64, 64, 33);
  k_dft_inv<<<gb(33024), 256, 0, stream>>>(IC256, IS256, 256, 129, 1, 256);
  k_dft_inv<<<gb(2112), 256, 0, stream>>>(IC64, IS64, 64, 33, 1, 64);
  k_dft_inv<<<gb(8256), 256, 0, stream>>>(ICse, ISse, 256, 129, 4, 64);   // irfft256 sampled at stride 4
  k_legendre<<<1, 32, 0, stream>>>(PTt, PWt);

  // psih_int = conj(rfft64(psi_int)): spatial [k][ho32][hi32][w64] -> [(k,ho,hi)][33]
  k_psi<<<gb(32ll * 32 * 64), 256, 0, stream>>>(SPAT, 32, 64, 32, 16.f * PI_F / 31.f,
                                                0, 32, 0, 9, 65536, 2048, 64);
  gemmL(stream, 9216, 33, 64, 1, 1, OP(SPAT, 0, 0, 64, 1), OP(C64, 0, 0, 33, 1),
        PintR, 0, 0, 33, 1, nullptr, 0, 0, 0);
  gemmL(stream, 9216, 33, 64, 1, 1, OP(SPAT, 0, 0, 64, 1), OP(S64, 0, 0, 33, 1),
        PintI, 0, 0, 33, 1, nullptr, 0, 0, 0);

  // psih_enc, chunked over k (3 at a time): spatial [kc][ho32][hi128][w256] -> [(k,ho,hi)][129]
  for (int k0 = 0; k0 < 9; k0 += 3) {
    k_psi<<<gb(32ll * 128 * 256), 256, 0, stream>>>(SPAT, 128, 256, 32, 4.f * PI_F / 31.f,
                                                    0, 32, k0, 3, 1048576, 32768, 256);
    gemmL(stream, 12288, 129, 256, 1, 1, OP(SPAT, 0, 0, 256, 1), OP(C256, 0, 0, 129, 1),
          PencR + (long long)k0 * 4096 * 129, 0, 0, 129, 1, nullptr, 0, 0, 0);
    gemmL(stream, 12288, 129, 256, 1, 1, OP(SPAT, 0, 0, 256, 1), OP(S256, 0, 0, 129, 1),
          PencI + (long long)k0 * 4096 * 129, 0, 0, 129, 1, nullptr, 0, 0, 0);
  }

  // ---- encoder: rfft256(x) ----
  gemmL(stream, 3072, 129, 256, 1, 1, OP(x, 0, 0, 256, 1), OP(C256, 0, 0, 129, 1),
        Xr256, 0, 0, 129, 1, nullptr, 0, 0, 0);
  gemmL(stream, 3072, 129, 256, 1, 1, OP(x, 0, 0, 256, 1), OP(S256, 0, 0, 129, 1),
        Xi256, 0, 0, 129, 1, nullptr, 0, 1, 0);
  // per-m complex contraction over hi: Zh[m][(k,ho)=288][(b,c)=24]
  gemmL(stream, 288, 24, 128, 129, 129, OP(PencR, 0, 1, 16512, 129), OP(Xr256, 0, 1, 129, 16512),
        ZeR, 0, 6912, 24, 1, nullptr, 0, 0, 0);
  gemmL(stream, 288, 24, 128, 129, 129, OP(PencI, 0, 1, 16512, 129), OP(Xi256, 0, 1, 129, 16512),
        ZeR, 0, 6912, 24, 1, nullptr, 1, 1, 0);
  gemmL(stream, 288, 24, 128, 129, 129, OP(PencR, 0, 1, 16512, 129), OP(Xi256, 0, 1, 129, 16512),
        ZeI, 0, 6912, 24, 1, nullptr, 0, 0, 0);
  gemmL(stream, 288, 24, 128, 129, 129, OP(PencI, 0, 1, 16512, 129), OP(Xr256, 0, 1, 129, 16512),
        ZeI, 0, 6912, 24, 1, nullptr, 1, 0, 0);
  // strided irfft (only wo multiples of 4) -> Zt[(k,ho,b,c)][wo64]
  gemmL(stream, 6912, 64, 129, 1, 1, OP(ZeR, 0, 0, 1, 6912), OP(ICse, 0, 0, 64, 1),
        Zt, 0, 0, 64, 1, nullptr, 0, 0, 0);
  gemmL(stream, 6912, 64, 129, 1, 1, OP(ZeI, 0, 0, 1, 6912), OP(ISse, 0, 0, 64, 1),
        Zt, 0, 0, 64, 1, nullptr, 1, 0, 0);
  // (k,ho,b,c,wo) -> [b][ho][wo][c][k]
  k_permute<<<gb(442368), 256, 0, stream>>>(Zt, ZP, 442368, 32, 8, 3, 64, 1, 1728, 55296, 9, 27);
  // encoder mix: NHWC activation [16384][256]
  gemmL(stream, 16384, 256, 27, 1, 1, OP(ZP, 0, 0, 27, 1), OP(encw, 0, 0, 1, 27),
        A0b, 0, 0, 256, 1, nullptr, 0, 0, 0);

  // ---- 4 interior blocks ----
  float *cur = A0b, *tmp = A1b, *nxt = A2b;
  for (int i = 0; i < 4; ++i) {
    int i2 = i / 2;
    // NHWC -> NCHW
    k_permute<<<gb(4194304), 256, 0, stream>>>(cur, XPb, 4194304, 32, 64, 256, 1,
                                               524288, 64, 1, 2048, 0);
    if ((i & 1) == 0) {
      // ---- local DISCO block ----
      gemmL(stream, 65536, 33, 64, 1, 1, OP(XPb, 0, 0, 64, 1), OP(C64, 0, 0, 33, 1),
            Xr64, 0, 0, 33, 1, nullptr, 0, 0, 0);
      gemmL(stream, 65536, 33, 64, 1, 1, OP(XPb, 0, 0, 64, 1), OP(S64, 0, 0, 33, 1),
            Xi64, 0, 0, 33, 1, nullptr, 0, 1, 0);
      for (int mc = 0; mc < 33; mc += 3) {
        // Zh chunk: [(b,c)][(k,ho)][mloc3]
        gemmL(stream, 288, 2048, 32, 3, 3, OP(PintR + mc, 0, 1, 1056, 33), OP(Xr64 + mc, 0, 1, 33, 1056),
              Z2R, 0, 1, 3, 864, nullptr, 0, 0, 0);
        gemmL(stream, 288, 2048, 32, 3, 3, OP(PintI + mc, 0, 1, 1056, 33), OP(Xi64 + mc, 0, 1, 33, 1056),
              Z2R, 0, 1, 3, 864, nullptr, 1, 1, 0);
        gemmL(stream, 288, 2048, 32, 3, 3, OP(PintR + mc, 0, 1, 1056, 33), OP(Xi64 + mc, 0, 1, 33, 1056),
              Z2I, 0, 1, 3, 864, nullptr, 0, 0, 0);
        gemmL(stream, 288, 2048, 32, 3, 3, OP(PintI + mc, 0, 1, 1056, 33), OP(Xr64 + mc, 0, 1, 33, 1056),
              Z2I, 0, 1, 3, 864, nullptr, 1, 0, 0);
        // channel mix over (c,k)=2304 -> Yh[b][o][ho][m33]
        gemmL(stream, 32, 256, 2304, 24, 3, OP(Z2R, 221184, 1, 3, 96),
              OP(lconv + (long long)i2 * 589824, 0, 0, 1, 2304),
              YhR + mc, 270336, 1, 33, 1056, nullptr, 0, 0, 0);
        gemmL(stream, 32, 256, 2304, 24, 3, OP(Z2I, 221184, 1, 3, 96),
              OP(lconv + (long long)i2 * 589824, 0, 0, 1, 2304),
              YhI + mc, 270336, 1, 33, 1056, nullptr, 0, 0, 0);
      }
      gemmL(stream, 65536, 64, 33, 1, 1, OP(YhR, 0, 0, 33, 1), OP(IC64, 0, 0, 64, 1),
            XPb, 0, 0, 64, 1, nullptr, 0, 0, 0);
      gemmL(stream, 65536, 64, 33, 1, 1, OP(YhI, 0, 0, 33, 1), OP(IS64, 0, 0, 64, 1),
            XPb, 0, 0, 64, 1, nullptr, 1, 0, 0);
    } else {
      // ---- global spectral (SHT) block ----
      gemmL(stream, 65536, 32, 64, 1, 1, OP(XPb, 0, 0, 64, 1), OP(C64, 0, 0, 33, 1),
            Xr64, 0, 0, 32, 1, nullptr, 0, 0, 0);
      gemmL(stream, 65536, 32, 64, 1, 1, OP(XPb, 0, 0, 64, 1), OP(S64, 0, 0, 33, 1),
            Xi64, 0, 0, 32, 1, nullptr, 0, 1, 0);
      // analysis: c[b][ch][l][m] (batch over m)
      gemmL(stream, 32, 2048, 32, 32, 32, OP(PWt, 0, 32, 1024, 1), OP(Xr64, 0, 1, 32, 1024),
            crB, 0, 1, 32, 1024, nullptr, 0, 0, 0);
      gemmL(stream, 32, 2048, 32, 32, 32, OP(PWt, 0, 32, 1024, 1), OP(Xi64, 0, 1, 32, 1024),
            ciB, 0, 1, 32, 1024, nullptr, 0, 0, 0);
      // per-l channel mix (batch over (b,l))
      gemmL(stream, 256, 32, 256, 256, 32, OP(gconv + (long long)i2 * 2097152, 0, 1, 8192, 32),
            OP(crB, 262144, 32, 1024, 1), c2R, 262144, 32, 1024, 1, nullptr, 0, 0, 0);
      gemmL(stream, 256, 32, 256, 256, 32, OP(gconv + (long long)i2 * 2097152, 0, 1, 8192, 32),
            OP(ciB, 262144, 32, 1024, 1), c2I, 262144, 32, 1024, 1, nullptr, 0, 0, 0);
      // synthesis: Xo[(b,o)][h][m] (batch over m)
      gemmL(stream, 32, 2048, 32, 32, 32, OP(PTt, 0, 32, 1, 1024), OP(c2R, 0, 1, 32, 1024),
            XoR, 0, 1, 32, 1024, nullptr, 0, 0, 0);
      gemmL(stream, 32, 2048, 32, 32, 32, OP(PTt, 0, 32, 1, 1024), OP(c2I, 0, 1, 32, 1024),
            XoI, 0, 1, 32, 1024, nullptr, 0, 0, 0);
      gemmL(stream, 65536, 64, 32, 1, 1, OP(XoR, 0, 0, 32, 1), OP(IC64, 0, 0, 64, 1),
            XPb, 0, 0, 64, 1, nullptr, 0, 0, 0);
      gemmL(stream, 65536, 64, 32, 1, 1, OP(XoI, 0, 0, 32, 1), OP(IS64, 0, 0, 64, 1),
            XPb, 0, 0, 64, 1, nullptr, 1, 0, 0);
    }
    // NCHW -> NHWC
    k_permute<<<gb(4194304), 256, 0, stream>>>(XPb, tmp, 4194304, 256, 32, 64, 1,
                                               524288, 1, 16384, 256, 0);
    // MLP + linear skip
    gemmL(stream, 16384, 512, 256, 1, 1, OP(tmp, 0, 0, 256, 1), OP(mw1 + (long long)i * 131072, 0, 0, 1, 256),
          H1b, 0, 0, 512, 1, mb1 + i * 512, 0, 0, 1);
    gemmL(stream, 16384, 256, 512, 1, 1, OP(H1b, 0, 0, 512, 1), OP(mw2 + (long long)i * 131072, 0, 0, 1, 512),
          nxt, 0, 0, 256, 1, mb2 + i * 256, 0, 0, 0);
    gemmL(stream, 16384, 256, 256, 1, 1, OP(cur, 0, 0, 256, 1), OP(skw + (long long)i * 65536, 0, 0, 1, 256),
          nxt, 0, 0, 256, 1, skb + i * 256, 1, 0, 0);
    float* t = cur; cur = nxt; nxt = t;
  }

  // ---- decoder ----
  // dec_w [o][c][k] -> [c][o][k]
  k_permute<<<gb(6912), 256, 0, stream>>>(decw, DWP, 6912, 256, 9, 1, 1, 9, 27, 1, 0, 0);
  // channel mix first (commutes with bilinear resample): v[pixel][ok=27]
  gemmL(stream, 16384, 27, 256, 1, 1, OP(cur, 0, 0, 256, 1), OP(DWP, 0, 0, 27, 1),
        vb, 0, 0, 27, 1, nullptr, 0, 0, 0);
  k_resample<<<gb(7077888), 256, 0, stream>>>(vb, ub);
  // rfft256 of u [(b,ok,hi)][256]
  gemmL(stream, 27648, 129, 256, 1, 1, OP(ub, 0, 0, 256, 1), OP(C256, 0, 0, 129, 1),
        UR, 0, 0, 129, 1, nullptr, 0, 0, 0);
  gemmL(stream, 27648, 129, 256, 1, 1, OP(ub, 0, 0, 256, 1), OP(S256, 0, 0, 129, 1),
        UI, 0, 0, 129, 1, nullptr, 0, 1, 0);
  // psih_dec built & applied in 8 latitude chunks of 16 output rows
  for (int hc = 0; hc < 8; ++hc) {
    k_psi<<<gb(16ll * 128 * 256), 256, 0, stream>>>(SPAT, 128, 256, 128, 4.f * PI_F / 31.f,
                                                    hc * 16, 16, 0, 9, 32768, 294912, 256);
    gemmL(stream, 18432, 129, 256, 1, 1, OP(SPAT, 0, 0, 256, 1), OP(C256, 0, 0, 129, 1),
          PDR, 0, 0, 129, 1, nullptr, 0, 0, 0);
    gemmL(stream, 18432, 129, 256, 1, 1, OP(SPAT, 0, 0, 256, 1), OP(S256, 0, 0, 129, 1),
          PDI, 0, 0, 129, 1, nullptr, 0, 0, 0);
    float* ydr = YdR + (long long)hc * 16 * 129;
    float* ydi = YdI + (long long)hc * 16 * 129;
    gemmL(stream, 16, 24, 1152, 129, 129, OP(PDR, 0, 1, 148608, 129), OP(UR, 0, 1, 129, 148608),
          ydr, 0, 1, 129, 16512, nullptr, 0, 0, 0);
    gemmL(stream, 16, 24, 1152, 129, 129, OP(PDI, 0, 1, 148608, 129), OP(UI, 0, 1, 129, 148608),
          ydr, 0, 1, 129, 16512, nullptr, 1, 1, 0);
    gemmL(stream, 16, 24, 1152, 129, 129, OP(PDR, 0, 1, 148608, 129), OP(UI, 0, 1, 129, 148608),
          ydi, 0, 1, 129, 16512, nullptr, 0, 0, 0);
    gemmL(stream, 16, 24, 1152, 129, 129, OP(PDI, 0, 1, 148608, 129), OP(UR, 0, 1, 129, 148608),
          ydi, 0, 1, 129, 16512, nullptr, 1, 0, 0);
  }
  // final irfft256 -> d_out [8,3,128,256]
  gemmL(stream, 3072, 256, 129, 1, 1, OP(YdR, 0, 0, 129, 1), OP(IC256, 0, 0, 256, 1),
        out, 0, 0, 256, 1, nullptr, 0, 0, 0);
  gemmL(stream, 3072, 256, 129, 1, 1, OP(YdI, 0, 0, 129, 1), OP(IS256, 0, 0, 256, 1),
        out, 0, 0, 256, 1, nullptr, 1, 0, 0);
}
